// DynamicRewireGNN_27702539059574
// MI455X (gfx1250) — compile-verified, matrix operand order checked by reading, not run on hardware
//
#include <hip/hip_runtime.h>
#include <cstdint>
#include <cstddef>

typedef __attribute__((ext_vector_type(16))) _Float16 v16h;
typedef __attribute__((ext_vector_type(8)))  _Float16 v8h;
typedef __attribute__((ext_vector_type(8)))  float    v8f;

#define HDIM 128

// ============================================================================
// Pre-swizzle W[KDIM x 128] (f32) into fragment-linear f16:
// element i: j=i&15 (half within lane), lane=(i>>4)&31, frag=i>>9, nt=frag&7,
// kt=frag>>3. Matches WMMA B-fragment layout (ISA 7.12.2) so GEMM blocks can
// stage LDS with a straight contiguous copy.
// ============================================================================
__global__ void swizzle_w_kernel(const float* __restrict__ W, _Float16* __restrict__ Wsw,
                                 int KDIM) {
    int i = blockIdx.x * blockDim.x + threadIdx.x;
    if (i >= KDIM * HDIM) return;
    const int j    = i & 15;
    const int lane = (i >> 4) & 31;
    const int frag = i >> 9;
    const int nt   = frag & 7;
    const int kt   = frag >> 3;
    const int sel8 = (lane & 16) ? 8 : 0;
    const int k    = kt * 32 + sel8 + ((j < 8) ? j : j + 8);
    const int n    = (nt << 4) + (lane & 15);
    Wsw[i] = (_Float16)W[k * HDIM + n];
}

// ============================================================================
// WMMA GEMM: out[M x 128] = maybe_relu( A[M x KDIM] @ W[KDIM x 128] + bias )
// GATHER=0: A = direct f32 rows.
// GATHER=1 (KDIM=256): row r = concat( Xh[idx0[r]], Xh[idx1[r]] ) in f16.
// Block = 128 threads = 4 waves; each wave computes 16 rows x 128 cols.
// Pre-swizzled weights are staged to LDS via GLOBAL_LOAD_ASYNC_TO_LDS_B128
// (ASYNCcnt-tracked), then consumed as 32B B-fragments.
// ============================================================================
template<int KDIM, int GATHER>
__global__ __launch_bounds__(128) void gemm128_kernel(
    const float*    __restrict__ A,
    const _Float16* __restrict__ Xh,
    const int*      __restrict__ idx0,
    const int*      __restrict__ idx1,
    const _Float16* __restrict__ Wsw,   // KDIM*128 halves, fragment-linear
    const float*    __restrict__ bias,
    float*          __restrict__ out,
    int M, int relu_flag)
{
    constexpr int KT = KDIM / 32;
    __shared__ _Float16 wlds[KDIM * HDIM] __attribute__((aligned(32)));

    const int tid = threadIdx.x;
    // Async copy (16B per lane per op): global fragment image -> LDS.
    {
        constexpr int NVEC = KDIM * HDIM / 8;  // 16-byte chunks
#pragma unroll
        for (int i = tid; i < NVEC; i += 128) {
            uint32_t ldsb = (uint32_t)(uintptr_t)(const void*)(wlds + i * 8);
            uint64_t ga   = (uint64_t)(uintptr_t)(const void*)(Wsw + i * 8);
            asm volatile("global_load_async_to_lds_b128 %0, %1, off"
                         :: "v"(ldsb), "v"(ga) : "memory");
        }
        asm volatile("s_wait_asynccnt 0x0" ::: "memory");
    }
    __syncthreads();

    const int wave = tid >> 5;
    const int lane = tid & 31;
    const int rowbase = blockIdx.x * 64 + wave * 16;
    if (rowbase >= M) return;                 // whole wave exits (M % 16 == 0)
    const int r    = rowbase + (lane & 15);
    const int sel8 = (lane & 16) ? 8 : 0;

    v8f acc[8];
#pragma unroll
    for (int nt = 0; nt < 8; ++nt)
#pragma unroll
        for (int j = 0; j < 8; ++j) acc[nt][j] = 0.0f;

    int i0 = 0, i1 = 0;
    if constexpr (GATHER) { i0 = idx0[r]; i1 = idx1[r]; }

#pragma unroll
    for (int kt = 0; kt < KT; ++kt) {
        v16h a;
        if constexpr (GATHER) {
            const int base = (kt < 4) ? i0 : i1;
            const _Float16* ap = Xh + (size_t)base * HDIM + ((kt & 3) * 32 + sel8);
            const v8h h0 = *(const v8h*)(ap);
            const v8h h1 = *(const v8h*)(ap + 16);
#pragma unroll
            for (int j = 0; j < 8; ++j) { a[j] = h0[j]; a[8 + j] = h1[j]; }
        } else {
            const float* ap = A + (size_t)r * KDIM + kt * 32 + sel8;
#pragma unroll
            for (int j = 0; j < 8; ++j) {
                a[j]     = (_Float16)ap[j];
                a[8 + j] = (_Float16)ap[16 + j];
            }
        }
#pragma unroll
        for (int nt = 0; nt < 8; ++nt) {
            const v16h b = *(const v16h*)(wlds + ((kt * 8 + nt) * 512 + lane * 16));
            acc[nt] = __builtin_amdgcn_wmma_f32_16x16x32_f16(
                false, a, false, b, (short)0, acc[nt], false, false);
        }
    }

#pragma unroll
    for (int nt = 0; nt < 8; ++nt) {
        const int   n  = (nt << 4) + (lane & 15);
        const float bn = bias[n];
#pragma unroll
        for (int v = 0; v < 8; ++v) {
            const int rowm = rowbase + sel8 + v;
            float val = acc[nt][v] + bn;
            if (relu_flag) val = fmaxf(val, 0.0f);
            out[(size_t)rowm * HDIM + n] = val;
        }
    }
}

// ============================================================================
// Elementwise / scatter / reduction kernels
// ============================================================================
__global__ void init_x_kernel(const float* __restrict__ x, float* __restrict__ xb,
                              _Float16* __restrict__ xh, int n) {
    int i = blockIdx.x * blockDim.x + threadIdx.x;
    if (i < n) { float v = x[i]; xb[i] = v; xh[i] = (_Float16)v; }
}

__global__ void refresh_xh_kernel(const float* __restrict__ xb, _Float16* __restrict__ xh, int n) {
    int i = blockIdx.x * blockDim.x + threadIdx.x;
    if (i < n) xh[i] = (_Float16)xb[i];
}

__global__ void compute_h_kernel(const float* __restrict__ xb, const float* __restrict__ agg,
                                 const float* __restrict__ eps_arr, int layer,
                                 float* __restrict__ hbuf, int n) {
    int i = blockIdx.x * blockDim.x + threadIdx.x;
    if (i < n) {
        float eps = eps_arr[layer];
        hbuf[i] = (1.0f + eps) * xb[i] + agg[i];
    }
}

__global__ void add_relu_kernel(float* __restrict__ xb, const float* __restrict__ t,
                                _Float16* __restrict__ xh, int n) {
    int i = blockIdx.x * blockDim.x + threadIdx.x;
    if (i < n) {
        float v = xb[i] + fmaxf(t[i], 0.0f);
        xb[i] = v;
        xh[i] = (_Float16)v;
    }
}

__global__ void split_cand_kernel(const int* __restrict__ ec, int* __restrict__ c0,
                                  int* __restrict__ c1, int C) {
    int i = blockIdx.x * blockDim.x + threadIdx.x;
    if (i < C) { c0[i] = ec[2 * i]; c1[i] = ec[2 * i + 1]; }
}

// agg[dst[e]] += relu(xb[src[e]] + ee[e]) * (ew[e] * kw[e]?)
__global__ void scatter_msg_kernel(const float* __restrict__ xb, const float* __restrict__ ee,
                                   const int* __restrict__ src, const int* __restrict__ dst,
                                   const float* __restrict__ ew, const float* __restrict__ kw,
                                   float* __restrict__ agg, int E) {
    int gid = blockIdx.x * blockDim.x + threadIdx.x;
    int e = gid >> 5;
    if (e >= E) return;
    int n = (gid & 31) * 4;
    float w = ew[e];
    if (kw) w *= kw[e];
    int s = src[e], d = dst[e];
    const float* xp = xb + (size_t)s * HDIM + n;
    const float* ep = ee + (size_t)e * HDIM + n;
    float* ap = agg + (size_t)d * HDIM + n;
#pragma unroll
    for (int j = 0; j < 4; ++j)
        unsafeAtomicAdd(ap + j, fmaxf(xp[j] + ep[j], 0.0f) * w);
}

// agg[c1[e]] += relu(xb[c0[e]] + ee_b) * cw[e]
__global__ void scatter_cand_kernel(const float* __restrict__ xb, const float* __restrict__ eeb,
                                    const int* __restrict__ c0, const int* __restrict__ c1,
                                    const float* __restrict__ cw, float* __restrict__ agg, int C) {
    int gid = blockIdx.x * blockDim.x + threadIdx.x;
    int e = gid >> 5;
    if (e >= C) return;
    int n = (gid & 31) * 4;
    float w = cw[e];
    int s = c0[e], d = c1[e];
    const float* xp = xb + (size_t)s * HDIM + n;
    float* ap = agg + (size_t)d * HDIM + n;
#pragma unroll
    for (int j = 0; j < 4; ++j)
        unsafeAtomicAdd(ap + j, fmaxf(xp[j] + eeb[n + j], 0.0f) * w);
}

// out[row] = sigmoid( mean over ENS=2 of (t[row] @ w2 + b2) ); one wave per row
__global__ void dot_sigmoid_kernel(const float* __restrict__ t, const float* __restrict__ w2,
                                   const float* __restrict__ b2, float* __restrict__ out, int rows) {
    int wid  = (blockIdx.x * blockDim.x + threadIdx.x) >> 5;
    int lane = threadIdx.x & 31;
    if (wid >= rows) return;
    const float* tp = t + (size_t)wid * HDIM + lane * 4;
    float s = 0.0f;
#pragma unroll
    for (int j = 0; j < 4; ++j) {
        int k = lane * 4 + j;
        s += tp[j] * (w2[k * 2] + w2[k * 2 + 1]);
    }
#pragma unroll
    for (int m = 16; m >= 1; m >>= 1) s += __shfl_xor(s, m, 32);
    if (lane == 0) {
        float mean = 0.5f * (s + b2[0] + b2[1]);
        out[wid] = 1.0f / (1.0f + __expf(-mean));
    }
}

__global__ void pool_scatter_kernel(const float* __restrict__ xb, const int* __restrict__ batch,
                                    float* __restrict__ pool, int N) {
    int gid = blockIdx.x * blockDim.x + threadIdx.x;
    int i = gid >> 5;
    if (i >= N) return;
    int n = (gid & 31) * 4;
    int g = batch[i];
    const float* xp = xb + (size_t)i * HDIM + n;
    float* pp = pool + (size_t)g * HDIM + n;
#pragma unroll
    for (int j = 0; j < 4; ++j) unsafeAtomicAdd(pp + j, xp[j]);
}

__global__ void cnt_kernel(const int* __restrict__ batch, float* __restrict__ cnt, int N) {
    int i = blockIdx.x * blockDim.x + threadIdx.x;
    if (i < N) unsafeAtomicAdd(cnt + batch[i], 1.0f);
}

// out[g,c] = (pool[g] . out_w[:,c]) / max(cnt[g],1) + out_b[c]; one wave per (g,c)
__global__ void out_kernel(const float* __restrict__ pool, const float* __restrict__ cnt,
                           const float* __restrict__ ow, const float* __restrict__ ob,
                           float* __restrict__ out, int G, int NC) {
    int wid  = (blockIdx.x * blockDim.x + threadIdx.x) >> 5;
    int lane = threadIdx.x & 31;
    if (wid >= G * NC) return;
    int g = wid / NC, c = wid % NC;
    const float* pp = pool + (size_t)g * HDIM + lane * 4;
    float s = 0.0f;
#pragma unroll
    for (int j = 0; j < 4; ++j) s += pp[j] * ow[(lane * 4 + j) * NC + c];
#pragma unroll
    for (int m = 16; m >= 1; m >>= 1) s += __shfl_xor(s, m, 32);
    if (lane == 0) out[g * NC + c] = s / fmaxf(cnt[g], 1.0f) + ob[c];
}

// ============================================================================
extern "C" void kernel_launch(void* const* d_in, const int* in_sizes, int n_in,
                              void* d_out, int out_size, void* d_ws, size_t ws_size,
                              hipStream_t stream) {
    (void)n_in; (void)out_size; (void)ws_size;
    const int N = in_sizes[0] / HDIM;     // 20000
    const int E = in_sizes[3];            // 320000
    const int C = in_sizes[4] / 2;        // 200000
    const int G = 64, NC = 10, L = 2;

    const float* x        = (const float*)d_in[0];
    const int*   ei       = (const int*)  d_in[1];
    const float* eattr    = (const float*)d_in[2];
    const float* ew       = (const float*)d_in[3];
    const int*   ecand    = (const int*)  d_in[4];
    const int*   batch    = (const int*)  d_in[5];
    const float* conv_w1  = (const float*)d_in[6];
    const float* conv_b1  = (const float*)d_in[7];
    const float* conv_w2  = (const float*)d_in[8];
    const float* conv_b2  = (const float*)d_in[9];
    const float* conv_eps = (const float*)d_in[10];
    const float* ee_w     = (const float*)d_in[11];
    const float* ee_b     = (const float*)d_in[12];
    const float* add_w1   = (const float*)d_in[13];
    const float* add_b1   = (const float*)d_in[14];
    const float* add_w2   = (const float*)d_in[15];
    const float* add_b2   = (const float*)d_in[16];
    const float* del_w1   = (const float*)d_in[17];
    const float* del_b1   = (const float*)d_in[18];
    const float* del_w2   = (const float*)d_in[19];
    const float* del_b2   = (const float*)d_in[20];
    const float* int_w1   = (const float*)d_in[21];
    const float* int_b1   = (const float*)d_in[22];
    const float* int_w2   = (const float*)d_in[23];
    const float* int_b2   = (const float*)d_in[24];
    const float* int_eps  = (const float*)d_in[25];
    const float* out_w    = (const float*)d_in[26];
    const float* out_b    = (const float*)d_in[27];
    float* dout = (float*)d_out;

    const int* src = ei;
    const int* dst = ei + E;

    // ---- workspace carving ----
    uintptr_t p = (uintptr_t)d_ws;
    auto carve = [&p](size_t bytes) -> void* {
        void* r = (void*)p;
        p += (bytes + 255) & ~(size_t)255;
        return r;
    };
    float*    ee    = (float*)   carve((size_t)E * HDIM * 4);
    float*    xb    = (float*)   carve((size_t)N * HDIM * 4);
    _Float16* xh    = (_Float16*)carve((size_t)N * HDIM * 2);
    float*    agg   = (float*)   carve((size_t)N * HDIM * 4);
    float*    hbuf  = (float*)   carve((size_t)N * HDIM * 4);
    float*    tbig  = (float*)   carve((size_t)E * HDIM * 4);
    float*    candw = (float*)   carve((size_t)C * 4);
    float*    keepw = (float*)   carve((size_t)E * 4);
    int*      c0b   = (int*)     carve((size_t)C * 4);
    int*      c1b   = (int*)     carve((size_t)C * 4);
    float*    pool  = (float*)   carve((size_t)G * HDIM * 4);
    float*    cnt   = (float*)   carve((size_t)G * 4);
    // pre-swizzled f16 weight images (fragment-linear)
    const size_t SW128 = (size_t)128 * HDIM * 2;
    const size_t SW256 = (size_t)256 * HDIM * 2;
    _Float16* sw_eew = (_Float16*)carve(SW128);
    _Float16* sw_cw1[2], *sw_cw2[2], *sw_iw1[2], *sw_iw2[2], *sw_aw1[2], *sw_dw1[2];
    for (int l = 0; l < L; ++l) {
        sw_cw1[l] = (_Float16*)carve(SW128);
        sw_cw2[l] = (_Float16*)carve(SW128);
        sw_iw1[l] = (_Float16*)carve(SW128);
        sw_iw2[l] = (_Float16*)carve(SW128);
        sw_aw1[l] = (_Float16*)carve(SW256);
        sw_dw1[l] = (_Float16*)carve(SW256);
    }

    const int nNH = N * HDIM;
    const dim3 b256(256), b128(128);
    const int gNH   = (nNH + 255) / 256;
    const int gE32  = (E * 32 + 255) / 256;
    const int gC32  = (C * 32 + 255) / 256;
    const int gN32  = (N * 32 + 255) / 256;
    const int gGemN = (N + 63) / 64;
    const int gGemE = (E + 63) / 64;
    const int gGemC = (C + 63) / 64;
    const int gSw128 = (128 * HDIM + 255) / 256;
    const int gSw256 = (256 * HDIM + 255) / 256;

    // ---- weight pre-swizzle (once per call, tiny) ----
    swizzle_w_kernel<<<gSw128, b256, 0, stream>>>(ee_w, sw_eew, 128);
    for (int l = 0; l < L; ++l) {
        swizzle_w_kernel<<<gSw128, b256, 0, stream>>>(conv_w1 + (size_t)l * HDIM * HDIM, sw_cw1[l], 128);
        swizzle_w_kernel<<<gSw128, b256, 0, stream>>>(conv_w2 + (size_t)l * HDIM * HDIM, sw_cw2[l], 128);
        swizzle_w_kernel<<<gSw128, b256, 0, stream>>>(int_w1  + (size_t)l * HDIM * HDIM, sw_iw1[l], 128);
        swizzle_w_kernel<<<gSw128, b256, 0, stream>>>(int_w2  + (size_t)l * HDIM * HDIM, sw_iw2[l], 128);
        swizzle_w_kernel<<<gSw256, b256, 0, stream>>>(add_w1 + (size_t)l * 2 * HDIM * HDIM, sw_aw1[l], 256);
        swizzle_w_kernel<<<gSw256, b256, 0, stream>>>(del_w1 + (size_t)l * 2 * HDIM * HDIM, sw_dw1[l], 256);
    }

    // ---- setup ----
    split_cand_kernel<<<(C + 255) / 256, b256, 0, stream>>>(ecand, c0b, c1b, C);
    init_x_kernel<<<gNH, b256, 0, stream>>>(x, xb, xh, nNH);
    // edge_emb = edge_attr @ ee_w + ee_b
    gemm128_kernel<128, 0><<<gGemE, b128, 0, stream>>>(
        eattr, nullptr, nullptr, nullptr, sw_eew, ee_b, ee, E, 0);

    for (int l = 0; l < L; ++l) {
        // ---- GINE conv ----
        hipMemsetAsync(agg, 0, (size_t)nNH * 4, stream);
        scatter_msg_kernel<<<gE32, b256, 0, stream>>>(xb, ee, src, dst, ew, nullptr, agg, E);
        compute_h_kernel<<<gNH, b256, 0, stream>>>(xb, agg, conv_eps, l, hbuf, nNH);
        gemm128_kernel<128, 0><<<gGemN, b128, 0, stream>>>(
            hbuf, nullptr, nullptr, nullptr, sw_cw1[l], conv_b1 + l * HDIM, tbig, N, 1);
        gemm128_kernel<128, 0><<<gGemN, b128, 0, stream>>>(
            tbig, nullptr, nullptr, nullptr, sw_cw2[l], conv_b2 + l * HDIM, hbuf, N, 0);
        add_relu_kernel<<<gNH, b256, 0, stream>>>(xb, hbuf, xh, nNH);

        // ---- candidate scoring (K=256 gathered GEMM + fused small layer) ----
        gemm128_kernel<256, 1><<<gGemC, b128, 0, stream>>>(
            nullptr, xh, c0b, c1b, sw_aw1[l], add_b1 + l * HDIM, tbig, C, 1);
        dot_sigmoid_kernel<<<gC32, b256, 0, stream>>>(
            tbig, add_w2 + (size_t)l * HDIM * 2, add_b2 + l * 2, candw, C);

        // ---- delete scoring ----
        gemm128_kernel<256, 1><<<gGemE, b128, 0, stream>>>(
            nullptr, xh, src, dst, sw_dw1[l], del_b1 + l * HDIM, tbig, E, 1);
        dot_sigmoid_kernel<<<gE32, b256, 0, stream>>>(
            tbig, del_w2 + (size_t)l * HDIM * 2, del_b2 + l * 2, keepw, E);

        // ---- interaction aggregation ----
        hipMemsetAsync(agg, 0, (size_t)nNH * 4, stream);
        scatter_msg_kernel<<<gE32, b256, 0, stream>>>(xb, ee, src, dst, ew, keepw, agg, E);
        scatter_cand_kernel<<<gC32, b256, 0, stream>>>(xb, ee_b, c0b, c1b, candw, agg, C);
        compute_h_kernel<<<gNH, b256, 0, stream>>>(xb, agg, int_eps, l, hbuf, nNH);
        gemm128_kernel<128, 0><<<gGemN, b128, 0, stream>>>(
            hbuf, nullptr, nullptr, nullptr, sw_iw1[l], int_b1 + l * HDIM, tbig, N, 1);
        gemm128_kernel<128, 0><<<gGemN, b128, 0, stream>>>(
            tbig, nullptr, nullptr, nullptr, sw_iw2[l], int_b2 + l * HDIM, xb, N, 0);
        refresh_xh_kernel<<<gNH, b256, 0, stream>>>(xb, xh, nNH);
    }

    // ---- mean pooling + output head ----
    hipMemsetAsync(pool, 0, (size_t)G * HDIM * 4, stream);
    hipMemsetAsync(cnt, 0, (size_t)G * 4, stream);
    pool_scatter_kernel<<<gN32, b256, 0, stream>>>(xb, batch, pool, N);
    cnt_kernel<<<(N + 255) / 256, b256, 0, stream>>>(batch, cnt, N);
    out_kernel<<<(G * NC * 32 + 255) / 256, b256, 0, stream>>>(pool, cnt, out_w, out_b, dout, G, NC);
}
